// SelectiveSSMExpert_55301998903668
// MI455X (gfx1250) — compile-verified, matrix operand order checked by reading, not run on hardware
//
#include <hip/hip_runtime.h>
#include <hip/hip_bf16.h>
#include <cmath>

// ---------------------------------------------------------------------------
// SelectiveSSMExpert for MI455X (gfx1250, wave32, WMMA bf16 16x16x32)
//   All GEMMs: LDS double-buffered async pipeline (ASYNCcnt +
//              GLOBAL_LOAD_ASYNC_TO_LDS_B128), WMMA bf16 compute.
//   GEMM3:     fused 4-gate + SSM update epilogue.
// ---------------------------------------------------------------------------

typedef __attribute__((ext_vector_type(16))) __bf16 v16bf;
typedef __attribute__((ext_vector_type(8)))  float  v8f;
typedef __attribute__((ext_vector_type(4)))  __bf16 v4bf;
typedef int v4i_g __attribute__((vector_size(16)));   // GCC-vector int4

#define DIM_   2048
#define STATE_ 2048
#define SELH_  1024
#define BATCH_ 16384

// ---- WMMA / data-movement helpers -----------------------------------------

__device__ __forceinline__ v8f wmma_bf16(v16bf a, v16bf b, v8f c) {
    return __builtin_amdgcn_wmma_f32_16x16x32_bf16(false, a, false, b,
                                                   (short)0, c, false, false);
}

// 16B fragment half-loads. p must be 16B aligned.
__device__ __forceinline__ v16bf load_frag(const __bf16* p) {
    v16bf v;
    reinterpret_cast<float4*>(&v)[0] = *reinterpret_cast<const float4*>(p);
    reinterpret_cast<float4*>(&v)[1] = *reinterpret_cast<const float4*>(p + 16);
    return v;
}

// async 16B global -> LDS copy (gfx1250 GLOBAL_LOAD_ASYNC_TO_LDS_B128)
__device__ __forceinline__ void cp_async16(const __bf16* g, __bf16* l) {
#if __has_builtin(__builtin_amdgcn_global_load_async_to_lds_b128)
    __builtin_amdgcn_global_load_async_to_lds_b128(
        (__attribute__((address_space(1))) v4i_g*)(g),
        (__attribute__((address_space(3))) v4i_g*)(l), 0, 0);
#else
    *reinterpret_cast<float4*>(l) = *reinterpret_cast<const float4*>(g);
#endif
}

__device__ __forceinline__ void wait_async() {
#if __has_builtin(__builtin_amdgcn_s_wait_asynccnt)
    __builtin_amdgcn_s_wait_asynccnt(0);
#else
    asm volatile("s_wait_asynccnt 0x0" ::: "memory");
#endif
}

__device__ __forceinline__ float sigmoidf_(float x) {
    return 1.0f / (1.0f + __expf(-x));
}

// ---- f32 -> bf16 conversion (vectorized, grid-stride) ----------------------

__global__ void cvt_f32_bf16_kernel(const float* __restrict__ in,
                                    __bf16* __restrict__ out, int n4) {
    int i      = blockIdx.x * blockDim.x + threadIdx.x;
    int stride = gridDim.x * blockDim.x;
    for (; i < n4; i += stride) {
        float4 f = reinterpret_cast<const float4*>(in)[i];
        v4bf o;
        o.x = (__bf16)f.x; o.y = (__bf16)f.y;
        o.z = (__bf16)f.z; o.w = (__bf16)f.w;
        reinterpret_cast<v4bf*>(out)[i] = o;
    }
}

// ---- Generic bf16 GEMM with LDS double buffering ---------------------------
//   C[M,N] = A[M,K] @ W[N,K]^T
//   block tile 128x128, K-stage 32, 8 waves, wave tile 32x64.
//   LDS rows padded to 40 elems (80B = 20 dwords; 16B aligned, bank-clean).
//   EPI==0: f32 store to Cf.   EPI==1: silu -> bf16 to Cb.

template <int EPI>
__global__ __launch_bounds__(256) void gemm_bf16_lds_kernel(
    const __bf16* __restrict__ A, const __bf16* __restrict__ W,
    float* __restrict__ Cf, __bf16* __restrict__ Cb,
    int M, int N, int K)
{
    constexpr int RS   = 40;          // padded row stride (elements)
    constexpr int TILE = 128 * RS;    // one buffer (elements)
    __shared__ __bf16 smem[4 * TILE]; // A[2 buf] then B[2 buf]  (40 KB)
    __bf16* As = smem;
    __bf16* Bs = smem + 2 * TILE;

    const int t     = threadIdx.x;
    const int lane  = t & 31;
    const int wid   = t >> 5;         // 0..7
    const int waveM = wid >> 1;       // 0..3
    const int waveN = wid & 1;        // 0..1
    const int r     = lane & 15;
    const int hi    = lane >> 4;

    const int mBlock = blockIdx.y * 128;
    const int nBlock = blockIdx.x * 128;

    // Stage loaders: 128 rows x 4 x 16B chunks per tile -> 2 chunks/thread/tile
    const __bf16* gA[2];
    const __bf16* gB[2];
    int lT[2];
#pragma unroll
    for (int i = 0; i < 2; ++i) {
        int c   = t + i * 256;        // 0..511
        int row = c >> 2;
        int seg = c & 3;
        gA[i] = A + (size_t)(mBlock + row) * K + seg * 8;
        gB[i] = W + (size_t)(nBlock + row) * K + seg * 8;
        lT[i] = row * RS + seg * 8;
    }

    // Fragment LDS read offsets (elements)
    int aIdx[2], bIdx[4];
#pragma unroll
    for (int mt = 0; mt < 2; ++mt)
        aIdx[mt] = (waveM * 32 + mt * 16 + r) * RS + hi * 8;
#pragma unroll
    for (int nt = 0; nt < 4; ++nt)
        bIdx[nt] = (waveN * 64 + nt * 16 + r) * RS + hi * 8;

    v8f acc[2][4];
#pragma unroll
    for (int mt = 0; mt < 2; ++mt)
#pragma unroll
        for (int nt = 0; nt < 4; ++nt) acc[mt][nt] = (v8f)(0.0f);

    auto issue = [&](int k0, int buf) {
#pragma unroll
        for (int i = 0; i < 2; ++i)
            cp_async16(gA[i] + k0, As + buf * TILE + lT[i]);
#pragma unroll
        for (int i = 0; i < 2; ++i)
            cp_async16(gB[i] + k0, Bs + buf * TILE + lT[i]);
    };

    const int NS = K >> 5;            // K/32 stages
    int buf = 0;
    issue(0, 0);
    wait_async();
    __syncthreads();

    for (int s = 0; s < NS; ++s) {
        if (s + 1 < NS) issue((s + 1) << 5, buf ^ 1);

        const __bf16* Ab = As + buf * TILE;
        const __bf16* Bb = Bs + buf * TILE;
        v16bf af[2], bfr[4];
#pragma unroll
        for (int mt = 0; mt < 2; ++mt) af[mt] = load_frag(Ab + aIdx[mt]);
#pragma unroll
        for (int nt = 0; nt < 4; ++nt) bfr[nt] = load_frag(Bb + bIdx[nt]);
#pragma unroll
        for (int mt = 0; mt < 2; ++mt)
#pragma unroll
            for (int nt = 0; nt < 4; ++nt)
                acc[mt][nt] = wmma_bf16(af[mt], bfr[nt], acc[mt][nt]);

        wait_async();
        __syncthreads();
        buf ^= 1;
    }

    // Epilogue. C frag: elem rr, lane l: M = rr + (l>=16?8:0), N = l&15.
#pragma unroll
    for (int mt = 0; mt < 2; ++mt) {
#pragma unroll
        for (int nt = 0; nt < 4; ++nt) {
            const int rowB = mBlock + waveM * 32 + mt * 16 + (hi << 3);
            const int col  = nBlock + waveN * 64 + nt * 16 + r;
#pragma unroll
            for (int rr = 0; rr < 8; ++rr) {
                const size_t idx = (size_t)(rowB + rr) * N + col;
                float v = acc[mt][nt][rr];
                if (EPI == 0) {
                    Cf[idx] = v;
                } else {
                    Cb[idx] = (__bf16)(v * sigmoidf_(v));   // silu
                }
            }
        }
    }
}

// ---- Fused GEMM3 + gates + SSM update (LDS async pipeline) -----------------
// sel[:, g*2048 + j] for g=0..3 accumulated together for the same state col j.
// Block tile: 128 rows x 64 state cols x 4 gates; B-tile = 256 Wso rows.
// LDS: A 2x(128x40) + B 2x(256x40) bf16 = 60 KB.

__global__ __launch_bounds__(256) void gemm_gates_kernel(
    const __bf16* __restrict__ selh,          // [B, SELH_]
    const __bf16* __restrict__ Wso,           // [4*STATE_, SELH_]
    const float*  __restrict__ hprev,         // [B, STATE_]
    const float*  __restrict__ u,             // [B, STATE_]
    const float*  __restrict__ dvec,          // [STATE_]
    float*        __restrict__ hout,          // [B, STATE_] (d_out 2nd half)
    __bf16*       __restrict__ ybf)           // [B, STATE_]
{
    constexpr int K     = SELH_;
    constexpr int RS    = 40;
    constexpr int ATILE = 128 * RS;           // 5120 elems
    constexpr int BTILE = 256 * RS;           // 10240 elems
    __shared__ __bf16 smem[2 * ATILE + 2 * BTILE];   // 60 KB
    __bf16* As = smem;
    __bf16* Bs = smem + 2 * ATILE;

    const int t     = threadIdx.x;
    const int lane  = t & 31;
    const int wid   = t >> 5;
    const int waveM = wid >> 1;               // 0..3
    const int waveN = wid & 1;                // 0..1
    const int r     = lane & 15;
    const int hi    = lane >> 4;

    const int mBlock = blockIdx.y * 128;
    const int nBlock = blockIdx.x * 64;       // state column base of block

    // A loaders: 512 chunks -> 2/thread
    const __bf16* gA[2];
    int lA[2];
#pragma unroll
    for (int i = 0; i < 2; ++i) {
        int c   = t + i * 256;
        int row = c >> 2;
        int seg = c & 3;
        gA[i] = selh + (size_t)(mBlock + row) * K + seg * 8;
        lA[i] = row * RS + seg * 8;
    }
    // B loaders: 1024 chunks (256 LDS rows = 4 gates x 64 cols) -> 4/thread
    const __bf16* gB[4];
    int lB[4];
#pragma unroll
    for (int i = 0; i < 4; ++i) {
        int c    = t + i * 256;
        int lrow = c >> 2;                    // 0..255
        int seg  = c & 3;
        int g    = lrow >> 6;                 // gate
        int lcol = lrow & 63;                 // state col within block
        gB[i] = Wso + (size_t)(g * STATE_ + nBlock + lcol) * K + seg * 8;
        lB[i] = lrow * RS + seg * 8;
    }

    // Fragment LDS read offsets
    int aIdx[2], bIdx[4][2];
#pragma unroll
    for (int mt = 0; mt < 2; ++mt)
        aIdx[mt] = (waveM * 32 + mt * 16 + r) * RS + hi * 8;
#pragma unroll
    for (int g = 0; g < 4; ++g)
#pragma unroll
        for (int nt = 0; nt < 2; ++nt)
            bIdx[g][nt] = ((g << 6) + waveN * 32 + nt * 16 + r) * RS + hi * 8;

    v8f acc[4][2][2];
#pragma unroll
    for (int g = 0; g < 4; ++g)
#pragma unroll
        for (int mt = 0; mt < 2; ++mt)
#pragma unroll
            for (int nt = 0; nt < 2; ++nt) acc[g][mt][nt] = (v8f)(0.0f);

    auto issue = [&](int k0, int buf) {
#pragma unroll
        for (int i = 0; i < 2; ++i)
            cp_async16(gA[i] + k0, As + buf * ATILE + lA[i]);
#pragma unroll
        for (int i = 0; i < 4; ++i)
            cp_async16(gB[i] + k0, Bs + buf * BTILE + lB[i]);
    };

    const int NS = K >> 5;                    // 32 stages
    int buf = 0;
    issue(0, 0);
    wait_async();
    __syncthreads();

    for (int s = 0; s < NS; ++s) {
        if (s + 1 < NS) issue((s + 1) << 5, buf ^ 1);

        const __bf16* Ab = As + buf * ATILE;
        const __bf16* Bb = Bs + buf * BTILE;
        v16bf af[2];
#pragma unroll
        for (int mt = 0; mt < 2; ++mt) af[mt] = load_frag(Ab + aIdx[mt]);
#pragma unroll
        for (int g = 0; g < 4; ++g) {
            v16bf bfr[2];
#pragma unroll
            for (int nt = 0; nt < 2; ++nt) bfr[nt] = load_frag(Bb + bIdx[g][nt]);
#pragma unroll
            for (int mt = 0; mt < 2; ++mt)
#pragma unroll
                for (int nt = 0; nt < 2; ++nt)
                    acc[g][mt][nt] = wmma_bf16(af[mt], bfr[nt], acc[g][mt][nt]);
        }

        wait_async();
        __syncthreads();
        buf ^= 1;
    }

    // Gate + SSM epilogue
#pragma unroll
    for (int mt = 0; mt < 2; ++mt) {
#pragma unroll
        for (int nt = 0; nt < 2; ++nt) {
            const int rowB = mBlock + waveM * 32 + mt * 16 + (hi << 3);
            const int j    = nBlock + waveN * 32 + nt * 16 + r;  // state col
            const float dj = dvec[j];
#pragma unroll
            for (int rr = 0; rr < 8; ++rr) {
                const size_t idx = (size_t)(rowB + rr) * STATE_ + j;
                float a  = sigmoidf_(acc[0][mt][nt][rr]);
                float b  = tanhf(acc[1][mt][nt][rr]);
                float c  = tanhf(acc[2][mt][nt][rr]);
                float dd = sigmoidf_(acc[3][mt][nt][rr]);
                float up = u[idx];
                float hp = hprev[idx];
                float h  = a * hp + b * up;
                float y  = c * h + dj * dd * up;
                hout[idx] = h;
                ybf[idx]  = (__bf16)y;
            }
        }
    }
}

// ---------------------------------------------------------------------------

extern "C" void kernel_launch(void* const* d_in, const int* in_sizes, int n_in,
                              void* d_out, int out_size, void* d_ws, size_t ws_size,
                              hipStream_t stream) {
    const float* x      = (const float*)d_in[0];   // [B, DIM]
    const float* hprev  = (const float*)d_in[1];   // [B, STATE]
    const float* W_in   = (const float*)d_in[2];   // [STATE, DIM]
    const float* W_si   = (const float*)d_in[3];   // [SELH, DIM]
    const float* W_so   = (const float*)d_in[4];   // [4*STATE, SELH]
    const float* W_out  = (const float*)d_in[5];   // [DIM, STATE]
    const float* dvec   = (const float*)d_in[6];   // [STATE]

    float* out_ptr = (float*)d_out;                         // [B, DIM]
    float* h_ptr   = (float*)d_out + (size_t)BATCH_ * DIM_; // [B, STATE]

    // ---- workspace layout ----
    char* ws = (char*)d_ws;
    size_t off = 0;
    auto take = [&](size_t bytes) {
        size_t r = off;
        off = (off + bytes + 255) & ~(size_t)255;
        return r;
    };
    __bf16* x_bf   = (__bf16*)(ws + take((size_t)BATCH_ * DIM_   * 2));
    __bf16* win_bf = (__bf16*)(ws + take((size_t)STATE_ * DIM_   * 2));
    __bf16* wsi_bf = (__bf16*)(ws + take((size_t)SELH_  * DIM_   * 2));
    __bf16* wso_bf = (__bf16*)(ws + take((size_t)4 * STATE_ * SELH_ * 2));
    __bf16* wou_bf = (__bf16*)(ws + take((size_t)DIM_   * STATE_ * 2));
    float*  u_f    = (float*) (ws + take((size_t)BATCH_ * STATE_ * 4));
    __bf16* selh   = (__bf16*)(ws + take((size_t)BATCH_ * SELH_  * 2));
    __bf16* y_bf   = (__bf16*)(ws + take((size_t)BATCH_ * STATE_ * 2));
    (void)ws_size; (void)in_sizes; (void)n_in; (void)out_size;

    // ---- f32 -> bf16 conversions ----
    auto cvt = [&](const float* src, __bf16* dst, size_t n) {
        int n4 = (int)(n / 4);
        int blocks = (n4 + 255) / 256;
        if (blocks > 2048) blocks = 2048;
        cvt_f32_bf16_kernel<<<blocks, 256, 0, stream>>>(src, dst, n4);
    };
    cvt(x,     x_bf,   (size_t)BATCH_ * DIM_);
    cvt(W_in,  win_bf, (size_t)STATE_ * DIM_);
    cvt(W_si,  wsi_bf, (size_t)SELH_  * DIM_);
    cvt(W_so,  wso_bf, (size_t)4 * STATE_ * SELH_);
    cvt(W_out, wou_bf, (size_t)DIM_   * STATE_);

    // ---- GEMM1: u = x @ W_in^T  [B, STATE] f32 ----
    gemm_bf16_lds_kernel<0><<<dim3(STATE_ / 128, BATCH_ / 128), 256, 0, stream>>>(
        x_bf, win_bf, u_f, nullptr, BATCH_, STATE_, DIM_);

    // ---- GEMM2: sel_h = silu(x @ W_sel_in^T)  [B, SELH] bf16 ----
    gemm_bf16_lds_kernel<1><<<dim3(SELH_ / 128, BATCH_ / 128), 256, 0, stream>>>(
        x_bf, wsi_bf, nullptr, selh, BATCH_, SELH_, DIM_);

    // ---- GEMM3 fused gates + SSM: h_t -> d_out, y -> y_bf ----
    gemm_gates_kernel<<<dim3(STATE_ / 64, BATCH_ / 128), 256, 0, stream>>>(
        selh, wso_bf, hprev, u_f, dvec, h_ptr, y_bf);

    // ---- GEMM4: out = y @ W_out^T  [B, DIM] f32 ----
    gemm_bf16_lds_kernel<0><<<dim3(DIM_ / 128, BATCH_ / 128), 256, 0, stream>>>(
        y_bf, wou_bf, out_ptr, nullptr, BATCH_, DIM_, STATE_);
}
